// Cellsort_Simulator_63694365000315
// MI455X (gfx1250) — compile-verified, matrix operand order checked by reading, not run on hardware
//
#include <hip/hip_runtime.h>
#include <hip/hip_bf16.h>

// ---------------------------------------------------------------------------
// Cellsort simulator, CDNA5 (gfx1250, wave32).
//
// Shapes: B=8, H=W=128, N=64 cells, T=4 types, E=16.
// Outputs concatenated in d_out (float):
//   probs        [B,2,H,W,N]  = 16,777,216 floats   (offset 0)
//   pred_disc    [B,2,H,W]    =    262,144 floats   (offset 16,777,216)
//   type_of_id   [B,N]        =        512 floats   (offset 17,039,360)
//   adj          [B,N,N]      =     32,768 floats   (offset 17,039,872)
//
// Math collapse: the per-pixel constant (type weight + bias) cancels inside
// both softmaxes, so only a = W_enc[0,:]·W_fin matters.  Channel-1 is the
// einsum probs_id[pix,64] @ type_scatter[64,64]; implemented with real
// v_wmma_f32_16x16x32_f16 (16-pixel M tiles, K=64 in two steps, N=64 in four
// 16-wide tiles).  68 MB of write-once output -> non-temporal stores; the
// kernel is HBM-store bound (~3 us at 23.3 TB/s), WMMA hides under stores.
// ---------------------------------------------------------------------------

typedef __attribute__((ext_vector_type(16))) _Float16 v16h;
typedef __attribute__((ext_vector_type(8)))  float    v8f;
typedef __attribute__((ext_vector_type(2)))  float    v2f;   // clang vector: OK for NT store

#define NB   8
#define HW   16384            // 128*128 pixels per batch
#define NC   64               // cells
#define EPSF 1e-6f

#define PROBS_ELEMS (2u*NC*HW*NB)                    // 16,777,216
#define DISC_OFF    ((size_t)PROBS_ELEMS)            // 16,777,216
#define TOI_OFF     (DISC_OFF + (size_t)2*HW*NB)     // 17,039,360
#define ADJ_OFF     (TOI_OFF + (size_t)NC*NB)        // 17,039,872

// ---------------------------------------------------------------------------
// Kernel 1: per-batch cell statistics -> COMs, adjacency, type_of_id, scalar a
// One workgroup per batch; LDS atomics over the 64 cell bins.
// ---------------------------------------------------------------------------
__global__ __launch_bounds__(256) void cellsort_reduce(
    const int* __restrict__ ids, const int* __restrict__ types,
    const float* __restrict__ Wenc, const float* __restrict__ Wfin,
    float* __restrict__ out, int* __restrict__ toi_ws, float* __restrict__ aPtr)
{
    __shared__ unsigned s_mass[NC], s_sh[NC], s_sw[NC], s_mt[NC];
    __shared__ float    s_ch[NC], s_cw[NC];
    __shared__ int      s_med;

    const int b   = blockIdx.x;
    const int tid = threadIdx.x;

    if (tid < NC) { s_mass[tid] = 0u; s_sh[tid] = 0u; s_sw[tid] = 0u; s_mt[tid] = 0u; }
    __syncthreads();

    for (int p = tid; p < HW; p += 256) {
        const int idx = b * HW + p;
        const int id  = ids[idx];
        const unsigned t = (unsigned)types[idx];
        atomicAdd(&s_mass[id], 1u);
        atomicAdd(&s_sh[id], (unsigned)(p >> 7));    // h
        atomicAdd(&s_sw[id], (unsigned)(p & 127));   // w
        atomicMax(&s_mt[id], t);
    }
    __syncthreads();

    if (tid < NC) {
        const float mass = (float)s_mass[tid];
        const float safe = fmaxf(mass, 1.0f);
        s_ch[tid] = (float)s_sh[tid] / safe;
        s_cw[tid] = (float)s_sw[tid] / safe;
        const int toi = (int)s_mt[tid];              // empty cell -> 0 (ref semantics)
        toi_ws[b * NC + tid]          = toi;
        out[TOI_OFF + b * NC + tid]   = (float)toi;
    }
    __syncthreads();

    if (tid == 0) {
        unsigned best = s_mass[0]; int bi = 0;       // first-max like jnp.argmax
        for (int k = 1; k < NC; ++k) if (s_mass[k] > best) { best = s_mass[k]; bi = k; }
        s_med = bi;
        if (b == 0) {                                 // a = W_enc[0,:] . W_fin
            float acc = 0.0f;
            for (int e = 0; e < 16; ++e) acc += Wenc[e] * Wfin[e];
            aPtr[0] = acc;
        }
    }
    __syncthreads();

    const int med = s_med;
    for (int q = tid; q < NC * NC; q += 256) {
        const int i = q >> 6, j = q & 63;
        const bool badi = (i == med) || (s_mass[i] == 0u);
        const bool badj = (j == med) || (s_mass[j] == 0u);
        const float dh = s_ch[i] - s_ch[j];
        const float dw = s_cw[i] - s_cw[j];
        const bool e = (sqrtf(dh * dh + dw * dw) <= 15.0f) && !badi && !badj;
        out[ADJ_OFF + (size_t)b * (NC * NC) + q] = e ? 1.0f : 0.0f;
    }
}

// ---------------------------------------------------------------------------
// Kernel 2: main streaming kernel. 1024 blocks x 256 threads (8 wave32s).
// Each wave owns a 16-pixel WMMA tile: channel-1 probs via WMMA f16->f32,
// channel-0 probs + pred_disc in closed form. All stores non-temporal.
// ---------------------------------------------------------------------------
__global__ __launch_bounds__(256) void cellsort_main(
    const int* __restrict__ ids, const int* __restrict__ toi_ws,
    const float* __restrict__ aPtr, float* __restrict__ out)
{
    __shared__ int s_toi[NC];
    const int b         = blockIdx.x >> 7;       // 128 blocks per batch
    const int pixelBase = blockIdx.x << 7;       // 128 pixels per block
    if (threadIdx.x < NC) s_toi[threadIdx.x] = toi_ws[b * NC + threadIdx.x];
    __syncthreads();

    const float a    = aPtr[0];
    const int  lane  = threadIdx.x & 31;
    const int  wave  = threadIdx.x >> 5;
    const int  tile0 = pixelBase + wave * 16;    // first of 16 pixels for this wave
    const int  m     = lane & 15;
    const int  pixel = tile0 + m;
    const int  id_m  = ids[pixel];               // lanes 16..31 duplicate lanes 0..15

    // Max-subtracted softmax over {c+a at n==id, c elsewhere}: c cancels.
    const float amx = fmaxf(a, 0.0f);
    const float e1  = __expf(a - amx);
    const float e0  = __expf(-amx);
    const float dn  = e1 + 63.0f * e0;
    const float p_match = e1 / dn;
    const float p_other = e0 / dn;
    const float inv_rs  = 1.0f / (p_match + 63.0f * p_other + 64.0f * EPSF);

    // --- A fragments (16x32 f16, M x K). lane L: row M=L&15,
    //     elem e -> K = kb + (e<8 ? e : e+8), kb = 8*(L>=16). Two K chunks.
    const _Float16 hm = (_Float16)p_match, ho = (_Float16)p_other;
    const int kb = (lane >= 16) ? 8 : 0;
    v16h A0, A1;
#pragma unroll
    for (int e = 0; e < 16; ++e) {
        const int K = kb + (e < 8 ? e : e + 8);
        A0[e] = (K == id_m)      ? hm : ho;
        A1[e] = (K + 32 == id_m) ? hm : ho;
    }

    // --- B = type_scatter (K x N, 0/1 exact in f16), symmetric column layout;
    //     four 16-wide N tiles, two K chunks each. Accumulate in f32.
    const int nlo = lane & 15;
    v8f acc[4];
#pragma unroll
    for (int j = 0; j < 4; ++j) {
        const int n = 16 * j + nlo;
        v16h B0, B1;
#pragma unroll
        for (int e = 0; e < 16; ++e) {
            const int K = kb + (e < 8 ? e : e + 8);
            B0[e] = (s_toi[K]      == n) ? (_Float16)1.0f : (_Float16)0.0f;
            B1[e] = (s_toi[K + 32] == n) ? (_Float16)1.0f : (_Float16)0.0f;
        }
        v8f c = {};
        c = __builtin_amdgcn_wmma_f32_16x16x32_f16(false, A0, false, B0,
                                                   (short)0, c, false, false);
        c = __builtin_amdgcn_wmma_f32_16x16x32_f16(false, A1, false, B1,
                                                   (short)0, c, false, false);
        acc[j] = c;
    }

    // --- Channel 0: [b,0,h,w,:] row index = (pixel + b*HW)*64 + n.
    const float c0m = p_match + EPSF, c0o = p_other + EPSF;
#pragma unroll 1
    for (int m2 = 0; m2 < 16; ++m2) {
        const int   idm2 = __shfl(id_m, m2, 32);
        const size_t row = ((size_t)(tile0 + m2) + (size_t)b * HW) * NC;
        const int n0 = lane * 2;
        v2f v;
        v.x = (n0     == idm2) ? c0m : c0o;
        v.y = (n0 + 1 == idm2) ? c0m : c0o;
        __builtin_nontemporal_store(v, reinterpret_cast<v2f*>(out + row) + lane);
    }

    // --- Channel 1: softmax(log(x+eps)) == (x+eps)/sum; per-batch constant sum.
    //     C/D layout: lane L elem r -> M = r + 8*(L>=16), N = 16j + (L&15).
    const int mhi = (lane >= 16) ? 8 : 0;
#pragma unroll
    for (int j = 0; j < 4; ++j) {
#pragma unroll
        for (int r = 0; r < 8; ++r) {
            const int   m2  = r + mhi;
            const size_t off = ((size_t)(tile0 + m2) + (size_t)b * HW + HW) * NC
                               + (16 * j + nlo);
            const float val = (acc[j][r] + EPSF) * inv_rs + EPSF;
            __builtin_nontemporal_store(val, out + off);
        }
    }

    // --- pred_disc: argmax over n of {c+a at id, c elsewhere}; first-max rule.
    if (lane < 16) {
        const int amax = (a > 0.0f) ? id_m : ((a < 0.0f && id_m == 0) ? 1 : 0);
        const int td   = s_toi[amax];
        const size_t p0 = DISC_OFF + (size_t)pixel + (size_t)b * HW;
        __builtin_nontemporal_store((float)amax, out + p0);
        __builtin_nontemporal_store((float)td,   out + p0 + HW);
    }
}

// ---------------------------------------------------------------------------
extern "C" void kernel_launch(void* const* d_in, const int* in_sizes, int n_in,
                              void* d_out, int out_size, void* d_ws, size_t ws_size,
                              hipStream_t stream) {
    const int*   ids  = (const int*)d_in[0];    // cell_ids   [8,1,128,128]
    const int*   typ  = (const int*)d_in[1];    // cell_types [8,1,128,128]
    const float* Wenc = (const float*)d_in[2];  // [5,16] row-major; row 0 used
    const float* Wfin = (const float*)d_in[4];  // [16]
    // b_enc (d_in[3]) and b_fin (d_in[5]) cancel inside both softmaxes.

    float* out    = (float*)d_out;
    int*   toi_ws = (int*)d_ws;                               // [8][64] int
    float* aPtr   = (float*)((char*)d_ws + NB * NC * sizeof(int));

    cellsort_reduce<<<NB, 256, 0, stream>>>(ids, typ, Wenc, Wfin, out, toi_ws, aPtr);
    cellsort_main<<<1024, 256, 0, stream>>>(ids, toi_ws, aPtr, out);
}
